// SimplifiedGraphAtten_regression_58866821759107
// MI455X (gfx1250) — compile-verified, matrix operand order checked by reading, not run on hardware
//
#include <hip/hip_runtime.h>
#include <hip/hip_bf16.h>
#include <math.h>

// ---------------------------------------------------------------------------
// Types for CDNA5 WMMA (wave32, 16x16x32 bf16 -> f32 accumulate)
// ---------------------------------------------------------------------------
typedef __attribute__((ext_vector_type(16))) __bf16 v16bf;
typedef __attribute__((ext_vector_type(8)))  float  v8f;
typedef unsigned short ushort_t;

#define HEADS 4
#define NTF   2000
#define NCRE  50000
#define NTGT  20000
#define ETT   30000
#define ETC   300000
#define ECC   500000
#define ECT   150000
#define ECTG  300000

__device__ __forceinline__ ushort_t f2bf(float x) {
  union { float f; unsigned u; } c; c.f = x;
  unsigned r = c.u + 0x7FFFu + ((c.u >> 16) & 1u);   // round-to-nearest-even
  return (ushort_t)(r >> 16);
}

// Fragment: 16 bf16 = 8 VGPRs; filled with two aligned 16-byte loads.
union Frag16 { v16bf v; uint4 q[2]; };

__device__ __forceinline__ void load_frag(const ushort_t* __restrict__ row, int ka, Frag16& f) {
  f.q[0] = *reinterpret_cast<const uint4*>(row + ka);        // K = ka .. ka+7   -> VGPR 0-3
  f.q[1] = *reinterpret_cast<const uint4*>(row + ka + 16);   // K = ka+16..ka+23 -> VGPR 4-7
}

// ---------------------------------------------------------------------------
// Conversion kernels (run once per matrix, amortized over all tile reuse)
// ---------------------------------------------------------------------------
__global__ void convert_bf16_kernel(const float* __restrict__ in, ushort_t* __restrict__ out, int n) {
  int i = blockIdx.x * blockDim.x + threadIdx.x;
  if (i < n) out[i] = f2bf(in[i]);
}

// Wt[n*K + k] = bf16( W[k*N + n] )  -- transpose so B fragments are K-contiguous
__global__ void transpose_bf16_kernel(const float* __restrict__ W, ushort_t* __restrict__ Wt,
                                      int K, int N) {
  int i = blockIdx.x * blockDim.x + threadIdx.x;
  if (i >= K * N) return;
  int n = i / K, k = i % K;
  Wt[i] = f2bf(W[(size_t)k * N + n]);
}

// ---------------------------------------------------------------------------
// GEMM: Out[M,N] = op( A[M,K](bf16) @ Bt[N,K](bf16)^T [+ bias] [+ Cin] )
//   MODE 0: Out = acc    MODE 1: Out = relu(acc)    MODE 2: Out += relu(acc)
// One wave computes a 16x32 strip (2 N-tiles, shared A fragment);
// K-loop unrolled x2 (K % 64 == 0 at all call sites). 8 waves / block.
// Fragment layouts per CDNA5 ISA 7.12.2; all fragment loads are b128.
// ---------------------------------------------------------------------------
template <int MODE, bool HAS_BIAS, bool HAS_CIN>
__global__ __launch_bounds__(256) void gemm_wmma_bf16(
    const ushort_t* __restrict__ A, const ushort_t* __restrict__ Bt,
    const float* __restrict__ bias, const float* __restrict__ Cin,
    float* __restrict__ Out, int M, int N, int K)
{
  const int wave = threadIdx.x >> 5;
  const int lane = threadIdx.x & 31;
  const int tilesN2 = N >> 5;                // strips of 32 columns
  const int tilesM  = M >> 4;
  const int t = blockIdx.x * 8 + wave;
  if (t >= tilesM * tilesN2) return;         // wave-uniform: EXEC stays all-ones
  const int tm = t / tilesN2;
  const int tn = (t % tilesN2) * 2;
  const int half = lane >> 4;                // K-half select (0/1)
  const int l15  = lane & 15;
  const ushort_t* Arow  = A  + (size_t)(tm * 16 + l15) * K;
  const int bcol0 = tn * 16 + l15;
  const int bcol1 = bcol0 + 16;
  const ushort_t* Brow0 = Bt + (size_t)bcol0 * K;
  const ushort_t* Brow1 = Bt + (size_t)bcol1 * K;

  v8f acc0 = {}, acc1 = {};
  for (int k0 = 0; k0 < K; k0 += 64) {
    __builtin_prefetch(Arow + k0 + 128, 0, 1);   // global_prefetch_b8
    const int ka = k0 + half * 8;
    Frag16 fa, fb0, fb1;
    load_frag(Arow,  ka, fa);
    load_frag(Brow0, ka, fb0);
    load_frag(Brow1, ka, fb1);
    acc0 = __builtin_amdgcn_wmma_f32_16x16x32_bf16(
        false, fa.v, false, fb0.v, (short)0, acc0, false, false);
    acc1 = __builtin_amdgcn_wmma_f32_16x16x32_bf16(
        false, fa.v, false, fb1.v, (short)0, acc1, false, false);
    Frag16 ga, gb0, gb1;
    load_frag(Arow,  ka + 32, ga);
    load_frag(Brow0, ka + 32, gb0);
    load_frag(Brow1, ka + 32, gb1);
    acc0 = __builtin_amdgcn_wmma_f32_16x16x32_bf16(
        false, ga.v, false, gb0.v, (short)0, acc0, false, false);
    acc1 = __builtin_amdgcn_wmma_f32_16x16x32_bf16(
        false, ga.v, false, gb1.v, (short)0, acc1, false, false);
  }

  // C/D layout: lanes 0-15 -> rows 0..7, lanes 16-31 -> rows 8..15; col = lane&15
  const float bv0 = HAS_BIAS ? bias[bcol0] : 0.0f;
  const float bv1 = HAS_BIAS ? bias[bcol1] : 0.0f;
#pragma unroll
  for (int r = 0; r < 8; ++r) {
    const int grow = tm * 16 + half * 8 + r;
    const size_t idx0 = (size_t)grow * N + bcol0;
    const size_t idx1 = (size_t)grow * N + bcol1;
    float v0 = acc0[r] + bv0;
    float v1 = acc1[r] + bv1;
    if (HAS_CIN) { v0 += Cin[idx0]; v1 += Cin[idx1]; }
    if (MODE == 0)      { Out[idx0] = v0;                        Out[idx1] = v1; }
    else if (MODE == 1) { Out[idx0] = v0 > 0.f ? v0 : 0.f;       Out[idx1] = v1 > 0.f ? v1 : 0.f; }
    else                { Out[idx0] += (v0 > 0.f ? v0 : 0.f);    Out[idx1] += (v1 > 0.f ? v1 : 0.f); }
  }
}

// ---------------------------------------------------------------------------
// Elementwise / edge kernels
// ---------------------------------------------------------------------------
__global__ void fill_kernel(float* p, float v, int n) {
  int i = blockIdx.x * blockDim.x + threadIdx.x;
  if (i < n) p[i] = v;
}

// out[i,h] = sum_d h[i,h,d] * a[h,d]
__global__ void attn_score_kernel(const float* __restrict__ h,
                                  const float* __restrict__ a,
                                  float* __restrict__ out, int n, int D) {
  int i = blockIdx.x * blockDim.x + threadIdx.x;
  if (i >= n * HEADS) return;
  int node = i / HEADS, hh = i % HEADS;
  const float* hp = h + (size_t)node * HEADS * D + hh * D;
  const float* ap = a + hh * D;
  float s = 0.f;
  for (int d = 0; d < D; ++d) s += hp[d] * ap[d];
  out[i] = s;
}

__device__ __forceinline__ float lrelu(float x) { return x > 0.f ? x : 0.2f * x; }

__device__ __forceinline__ void atomicMaxF(float* addr, float v) {
  if (v >= 0.f) atomicMax((int*)addr, __float_as_int(v));
  else          atomicMin((unsigned int*)addr, __float_as_uint(v));
}

__global__ void edge_max_kernel(const int* __restrict__ src, const int* __restrict__ dst,
                                const float* __restrict__ as_, const float* __restrict__ ad_,
                                float* __restrict__ m, int E) {
  int i = blockIdx.x * blockDim.x + threadIdx.x;
  if (i >= E * HEADS) return;
  int e = i / HEADS, hh = i % HEADS;
  int s = src[e], d = dst[e];
  float lg = lrelu(as_[s * HEADS + hh] + ad_[d * HEADS + hh]);
  atomicMaxF(&m[d * HEADS + hh], lg);
}

__global__ void edge_expsum_kernel(const int* __restrict__ src, const int* __restrict__ dst,
                                   const float* __restrict__ as_, const float* __restrict__ ad_,
                                   const float* __restrict__ m, float* __restrict__ sums, int E) {
  int i = blockIdx.x * blockDim.x + threadIdx.x;
  if (i >= E * HEADS) return;
  int e = i / HEADS, hh = i % HEADS;
  int s = src[e], d = dst[e];
  float lg = lrelu(as_[s * HEADS + hh] + ad_[d * HEADS + hh]);
  atomicAdd(&sums[d * HEADS + hh], __expf(lg - m[d * HEADS + hh]));
}

__global__ void edge_agg_kernel(const int* __restrict__ src, const int* __restrict__ dst,
                                const float* __restrict__ as_, const float* __restrict__ ad_,
                                const float* __restrict__ m, const float* __restrict__ sums,
                                const float* __restrict__ hs, float* __restrict__ out,
                                int E, int D) {
  int i = blockIdx.x * blockDim.x + threadIdx.x;
  if (i >= E * HEADS) return;
  int e = i / HEADS, hh = i % HEADS;
  int s = src[e], d = dst[e];
  float lg = lrelu(as_[s * HEADS + hh] + ad_[d * HEADS + hh]);
  float w = __expf(lg - m[d * HEADS + hh]) / (sums[d * HEADS + hh] + 1e-16f);
  const float* hp = hs + (size_t)s * HEADS * D + hh * D;
  float* op = out + (size_t)d * HEADS * D + hh * D;
  for (int dd = 0; dd < D; ++dd) atomicAdd(&op[dd], hp[dd] * w);
}

// out (+)= relu(in + bias)   mode 1: assign, mode 2: accumulate
__global__ void bias_act_kernel(const float* __restrict__ in, const float* __restrict__ bias,
                                float* __restrict__ out, int M, int N, int mode) {
  int i = blockIdx.x * blockDim.x + threadIdx.x;
  if (i >= M * N) return;
  int c = i % N;
  float v = in[i] + (bias ? bias[c] : 0.f);
  float r = v > 0.f ? v : 0.f;
  if (mode == 1) out[i] = r; else out[i] += r;
}

__global__ void gcn_deg_kernel(const int* __restrict__ dst, float* __restrict__ deg, int E) {
  int e = blockIdx.x * blockDim.x + threadIdx.x;
  if (e < E) atomicAdd(&deg[dst[e]], 1.0f);
}

// self-loop term + bias: out[i,c] = h[i,c]/deg[i] + bias[c]
__global__ void gcn_self_kernel(const float* __restrict__ h, const float* __restrict__ deg,
                                const float* __restrict__ bias, float* __restrict__ out,
                                int M, int N) {
  int i = blockIdx.x * blockDim.x + threadIdx.x;
  if (i >= M * N) return;
  int r = i / N, c = i % N;
  out[i] = h[i] / deg[r] + bias[c];
}

__global__ void gcn_edge_kernel(const int* __restrict__ src, const int* __restrict__ dst,
                                const float* __restrict__ h, const float* __restrict__ deg,
                                float* __restrict__ out, int E, int N) {
  int i = blockIdx.x * blockDim.x + threadIdx.x;
  if (i >= E * N) return;
  int e = i / N, c = i % N;
  int s = src[e], d = dst[e];
  float w = rsqrtf(deg[s]) * rsqrtf(deg[d]);
  atomicAdd(&out[(size_t)d * N + c], h[(size_t)s * N + c] * w);
}

// ---------------------------------------------------------------------------
// Host-side orchestration
// ---------------------------------------------------------------------------
static inline int cdiv(int a, int b) { return (a + b - 1) / b; }

static void gemm(hipStream_t st, const ushort_t* A, const ushort_t* Bt, const float* bias,
                 const float* Cin, float* Out, int M, int N, int K, int mode) {
  int tiles = (M / 16) * (N / 32);
  dim3 g(cdiv(tiles, 8)), b(256);
  if (mode == 0) {
    if (bias && !Cin)       gemm_wmma_bf16<0, true,  false><<<g, b, 0, st>>>(A, Bt, bias, Cin, Out, M, N, K);
    else if (!bias && !Cin) gemm_wmma_bf16<0, false, false><<<g, b, 0, st>>>(A, Bt, bias, Cin, Out, M, N, K);
    else if (bias && Cin)   gemm_wmma_bf16<0, true,  true ><<<g, b, 0, st>>>(A, Bt, bias, Cin, Out, M, N, K);
    else                    gemm_wmma_bf16<0, false, true ><<<g, b, 0, st>>>(A, Bt, bias, Cin, Out, M, N, K);
  } else if (mode == 1) {
    if (bias && !Cin)       gemm_wmma_bf16<1, true,  false><<<g, b, 0, st>>>(A, Bt, bias, Cin, Out, M, N, K);
    else if (!bias && !Cin) gemm_wmma_bf16<1, false, false><<<g, b, 0, st>>>(A, Bt, bias, Cin, Out, M, N, K);
    else if (bias && Cin)   gemm_wmma_bf16<1, true,  true ><<<g, b, 0, st>>>(A, Bt, bias, Cin, Out, M, N, K);
    else                    gemm_wmma_bf16<1, false, true ><<<g, b, 0, st>>>(A, Bt, bias, Cin, Out, M, N, K);
  } else {
    if (bias && !Cin)       gemm_wmma_bf16<2, true,  false><<<g, b, 0, st>>>(A, Bt, bias, Cin, Out, M, N, K);
    else if (!bias && !Cin) gemm_wmma_bf16<2, false, false><<<g, b, 0, st>>>(A, Bt, bias, Cin, Out, M, N, K);
    else if (bias && Cin)   gemm_wmma_bf16<2, true,  true ><<<g, b, 0, st>>>(A, Bt, bias, Cin, Out, M, N, K);
    else                    gemm_wmma_bf16<2, false, true ><<<g, b, 0, st>>>(A, Bt, bias, Cin, Out, M, N, K);
  }
}

struct GatP {
  const float *ad, *as, *b;       // attention vectors + bias (f32)
  const ushort_t *WdT, *WsT;      // pre-transposed bf16 weights [dout, din]
};

static void run_gat(hipStream_t st, const ushort_t* bxs, int Ns, const ushort_t* bxd, int Nd,
                    const int* ei, int E, const GatP& p, int din, int dout,
                    float* oacc, int accmode,
                    float* hs, float* hd, float* asb, float* adb,
                    float* mb, float* sb, float* tbuf) {
  const int D = dout / HEADS;
  gemm(st, bxs, p.WsT, nullptr, nullptr, hs, Ns, dout, din, 0);
  gemm(st, bxd, p.WdT, nullptr, nullptr, hd, Nd, dout, din, 0);
  attn_score_kernel<<<cdiv(Ns * HEADS, 256), 256, 0, st>>>(hs, p.as, asb, Ns, D);
  attn_score_kernel<<<cdiv(Nd * HEADS, 256), 256, 0, st>>>(hd, p.ad, adb, Nd, D);
  fill_kernel<<<cdiv(Nd * HEADS, 256), 256, 0, st>>>(mb, -INFINITY, Nd * HEADS);
  fill_kernel<<<cdiv(Nd * HEADS, 256), 256, 0, st>>>(sb, 0.f, Nd * HEADS);
  fill_kernel<<<cdiv(Nd * dout, 256), 256, 0, st>>>(tbuf, 0.f, Nd * dout);
  const int* src = ei;
  const int* dst = ei + E;
  edge_max_kernel<<<cdiv(E * HEADS, 256), 256, 0, st>>>(src, dst, asb, adb, mb, E);
  edge_expsum_kernel<<<cdiv(E * HEADS, 256), 256, 0, st>>>(src, dst, asb, adb, mb, sb, E);
  edge_agg_kernel<<<cdiv(E * HEADS, 256), 256, 0, st>>>(src, dst, asb, adb, mb, sb, hs, tbuf, E, D);
  bias_act_kernel<<<cdiv(Nd * dout, 256), 256, 0, st>>>(tbuf, p.b, oacc, Nd, dout, accmode);
}

extern "C" void kernel_launch(void* const* d_in, const int* in_sizes, int n_in,
                              void* d_out, int out_size, void* d_ws, size_t ws_size,
                              hipStream_t stream) {
  (void)in_sizes; (void)n_in; (void)out_size; (void)ws_size;
  const float* x_tf  = (const float*)d_in[0];
  const float* x_cre = (const float*)d_in[1];
  const float* x_tgt = (const float*)d_in[2];
  const int* ei_tt  = (const int*)d_in[3];
  const int* ei_tc  = (const int*)d_in[4];
  const int* ei_cc  = (const int*)d_in[5];
  const int* ei_ct  = (const int*)d_in[6];
  const int* ei_ctg = (const int*)d_in[7];

  // Params flattened in JAX pytree order: head(4x{W,b}), layer0, layer1, lin{W,b}.
  // Each layer (sorted keys): CRE_CRE, CRE_TF, CRE_Target, TF_CRE (each W_dst,W_src,a_dst,a_src,b),
  // then TF_TF {W,b}, sl_CRE {W_l,W_r,b}, sl_TF {W_l,W_r,b}.
  int pi = 8;
  const float* headWf[4]; const float* headB[4];
  for (int i = 0; i < 4; ++i) { headWf[i] = (const float*)d_in[pi++]; headB[i] = (const float*)d_in[pi++]; }
  struct LayerRaw {
    const float *W[13];                   // ccWd,ccWs, ctWd,ctWs, ctgWd,ctgWs, tcWd,tcWs, gW, scWl,scWr, stWl,stWr
    const float *cc_ad, *cc_as, *cc_b;
    const float *ct_ad, *ct_as, *ct_b;
    const float *ctg_ad, *ctg_as, *ctg_b;
    const float *tc_ad, *tc_as, *tc_b;
    const float *gb, *scb, *stb;
  } LR[2];
  for (int li = 0; li < 2; ++li) {
    const float** adp[4] = { &LR[li].cc_ad,  &LR[li].ct_ad,  &LR[li].ctg_ad,  &LR[li].tc_ad  };
    const float** asp[4] = { &LR[li].cc_as,  &LR[li].ct_as,  &LR[li].ctg_as,  &LR[li].tc_as  };
    const float** bp [4] = { &LR[li].cc_b,   &LR[li].ct_b,   &LR[li].ctg_b,   &LR[li].tc_b   };
    for (int g = 0; g < 4; ++g) {
      LR[li].W[2 * g]     = (const float*)d_in[pi++];   // W_dst
      LR[li].W[2 * g + 1] = (const float*)d_in[pi++];   // W_src
      *adp[g] = (const float*)d_in[pi++];
      *asp[g] = (const float*)d_in[pi++];
      *bp[g]  = (const float*)d_in[pi++];
    }
    LR[li].W[8]  = (const float*)d_in[pi++];  LR[li].gb  = (const float*)d_in[pi++];  // TF_TF
    LR[li].W[9]  = (const float*)d_in[pi++];  LR[li].W[10] = (const float*)d_in[pi++];
    LR[li].scb = (const float*)d_in[pi++];                                            // sl_CRE
    LR[li].W[11] = (const float*)d_in[pi++];  LR[li].W[12] = (const float*)d_in[pi++];
    LR[li].stb = (const float*)d_in[pi++];                                            // sl_TF
  }
  const float* linWf = (const float*)d_in[pi++];
  const float* linB  = (const float*)d_in[pi++];

  // Workspace carve (byte allocator, 256B aligned)
  char* base = (char*)d_ws;
  size_t off = 0;
  auto allocf = [&](size_t n) { float* p = (float*)(base + off); off = (off + n * 4 + 255) & ~(size_t)255; return p; };
  auto alloch = [&](size_t n) { ushort_t* p = (ushort_t*)(base + off); off = (off + n * 2 + 255) & ~(size_t)255; return p; };

  float* oA_tf  = allocf((size_t)NTF  * 256);
  float* oA_cre = allocf((size_t)NCRE * 256);
  float* oA_tg  = allocf((size_t)NTGT * 256);
  float* oB_tf  = allocf((size_t)NTF  * 256);
  float* oB_cre = allocf((size_t)NCRE * 256);
  float* oB_tg  = allocf((size_t)NTGT * 256);
  float* tmpS   = allocf((size_t)NCRE * 256);   // SAGE tmp / GCN h / head h1
  float* tbuf   = allocf((size_t)NCRE * 256);   // GAT/GCN aggregation buffer / head h2
  float* hs     = allocf((size_t)NCRE * 256);   // GAT src projection / head h3
  float* hd     = allocf((size_t)NCRE * 256);   // GAT dst projection
  float* asb    = allocf((size_t)NCRE * HEADS);
  float* adb    = allocf((size_t)NCRE * HEADS);
  float* mb     = allocf((size_t)NCRE * HEADS);
  float* sb     = allocf((size_t)NCRE * HEADS);
  float* deg    = allocf((size_t)NTF);
  ushort_t* bx_tf  = alloch((size_t)NTF  * 256);   // bf16 activation mirrors (A operands)
  ushort_t* bx_cre = alloch((size_t)NCRE * 256);
  ushort_t* bx_tg  = alloch((size_t)NTGT * 256);
  ushort_t* bx_scr = alloch((size_t)NCRE * 256);   // scratch for head A conversions

  // --- pre-convert + transpose all weights to bf16 [N,K] ---
  auto prepW = [&](const float* W, int K, int N) {
    ushort_t* p = alloch((size_t)K * N);
    transpose_bf16_kernel<<<cdiv(K * N, 256), 256, 0, stream>>>(W, p, K, N);
    return p;
  };
  auto aconv = [&](const float* f, ushort_t* h, size_t n) {
    convert_bf16_kernel<<<cdiv((int)n, 256), 256, 0, stream>>>(f, h, (int)n);
  };

  const int dins[2]  = { 256, 256 };
  const int douts[2] = { 256, 128 };
  const ushort_t* WT[2][13];
  for (int li = 0; li < 2; ++li)
    for (int wi = 0; wi < 13; ++wi)
      WT[li][wi] = prepW(LR[li].W[wi], dins[li], douts[li]);
  const ushort_t* linWT  = prepW(linWf, 128, 128);
  const ushort_t* headWT[4] = { prepW(headWf[0], 128, 128), prepW(headWf[1], 128, 256),
                                prepW(headWf[2], 256, 384), prepW(headWf[3], 384, 256) };

  const float* xtf = x_tf; const float* xcre = x_cre; const float* xtg = x_tgt;
  float* otf_b[2]  = { oA_tf,  oB_tf  };
  float* ocre_b[2] = { oA_cre, oB_cre };
  float* otg_b[2]  = { oA_tg,  oB_tg  };

  for (int li = 0; li < 2; ++li) {
    const int din = dins[li], dout = douts[li];
    float* otf = otf_b[li]; float* ocre = ocre_b[li]; float* otg = otg_b[li];

    // bf16 mirrors of this layer's input activations
    aconv(xtf,  bx_tf,  (size_t)NTF  * din);
    aconv(xcre, bx_cre, (size_t)NCRE * din);
    aconv(xtg,  bx_tg,  (size_t)NTGT * din);

    GatP pcc  = { LR[li].cc_ad,  LR[li].cc_as,  LR[li].cc_b,  WT[li][0], WT[li][1] };
    GatP pct  = { LR[li].ct_ad,  LR[li].ct_as,  LR[li].ct_b,  WT[li][2], WT[li][3] };
    GatP pctg = { LR[li].ctg_ad, LR[li].ctg_as, LR[li].ctg_b, WT[li][4], WT[li][5] };
    GatP ptc  = { LR[li].tc_ad,  LR[li].tc_as,  LR[li].tc_b,  WT[li][6], WT[li][7] };

    // SAGE self (mean over self-loop == x): o = relu(x@W_l + x@W_r + b)
    gemm(stream, bx_tf,  WT[li][11], LR[li].stb, nullptr, tmpS, NTF,  dout, din, 0);
    gemm(stream, bx_tf,  WT[li][12], nullptr,    tmpS,    otf,  NTF,  dout, din, 1);
    gemm(stream, bx_cre, WT[li][9],  LR[li].scb, nullptr, tmpS, NCRE, dout, din, 0);
    gemm(stream, bx_cre, WT[li][10], nullptr,    tmpS,    ocre, NCRE, dout, din, 1);

    // GCN TF->TF: o_tf += relu(gcn)
    fill_kernel<<<cdiv(NTF, 256), 256, 0, stream>>>(deg, 1.0f, NTF);   // self-loops
    gcn_deg_kernel<<<cdiv(ETT, 256), 256, 0, stream>>>(ei_tt + ETT, deg, ETT);
    gemm(stream, bx_tf, WT[li][8], nullptr, nullptr, tmpS, NTF, dout, din, 0);
    gcn_self_kernel<<<cdiv(NTF * dout, 256), 256, 0, stream>>>(tmpS, deg, LR[li].gb, tbuf, NTF, dout);
    gcn_edge_kernel<<<cdiv(ETT * dout, 256), 256, 0, stream>>>(ei_tt, ei_tt + ETT, tmpS, deg, tbuf, ETT, dout);
    bias_act_kernel<<<cdiv(NTF * dout, 256), 256, 0, stream>>>(tbuf, nullptr, otf, NTF, dout, 2);

    // GATs
    run_gat(stream, bx_tf,  NTF,  bx_cre, NCRE, ei_tc,  ETC,  ptc,  din, dout, ocre, 2, hs, hd, asb, adb, mb, sb, tbuf);
    run_gat(stream, bx_cre, NCRE, bx_cre, NCRE, ei_cc,  ECC,  pcc,  din, dout, ocre, 2, hs, hd, asb, adb, mb, sb, tbuf);
    run_gat(stream, bx_cre, NCRE, bx_tf,  NTF,  ei_ct,  ECT,  pct,  din, dout, otf,  2, hs, hd, asb, adb, mb, sb, tbuf);
    run_gat(stream, bx_cre, NCRE, bx_tg,  NTGT, ei_ctg, ECTG, pctg, din, dout, otg,  1, hs, hd, asb, adb, mb, sb, tbuf);

    xtf = otf; xcre = ocre; xtg = otg;
  }

  // Outputs: z_tf, z_cre, z_tgt, head(z_tgt), head(z_tf)
  float* out   = (float*)d_out;
  float* z_tf  = out;
  float* z_cre = out + (size_t)NTF * 128;
  float* z_tgt = out + (size_t)(NTF + NCRE) * 128;
  float* h_tgt = out + (size_t)(NTF + NCRE + NTGT) * 128;
  float* h_tf  = h_tgt + (size_t)NTGT * 256;

  aconv(xtf,  bx_tf,  (size_t)NTF  * 128);
  aconv(xcre, bx_cre, (size_t)NCRE * 128);
  aconv(xtg,  bx_tg,  (size_t)NTGT * 128);
  gemm(stream, bx_tf,  linWT, linB, nullptr, z_tf,  NTF,  128, 128, 0);
  gemm(stream, bx_cre, linWT, linB, nullptr, z_cre, NCRE, 128, 128, 0);
  gemm(stream, bx_tg,  linWT, linB, nullptr, z_tgt, NTGT, 128, 128, 0);

  // head(z_tgt): 128 -> 128 -> 256 -> 384 -> 256
  aconv(z_tgt, bx_scr, (size_t)NTGT * 128);
  gemm(stream, bx_scr, headWT[0], headB[0], nullptr, tmpS,  NTGT, 128, 128, 1);
  aconv(tmpS,  bx_scr, (size_t)NTGT * 128);
  gemm(stream, bx_scr, headWT[1], headB[1], nullptr, tbuf,  NTGT, 256, 128, 1);
  aconv(tbuf,  bx_scr, (size_t)NTGT * 256);
  gemm(stream, bx_scr, headWT[2], headB[2], nullptr, hs,    NTGT, 384, 256, 1);
  aconv(hs,    bx_scr, (size_t)NTGT * 384);
  gemm(stream, bx_scr, headWT[3], headB[3], nullptr, h_tgt, NTGT, 256, 384, 0);

  // head(z_tf)
  aconv(z_tf,  bx_scr, (size_t)NTF * 128);
  gemm(stream, bx_scr, headWT[0], headB[0], nullptr, tmpS,  NTF, 128, 128, 1);
  aconv(tmpS,  bx_scr, (size_t)NTF * 128);
  gemm(stream, bx_scr, headWT[1], headB[1], nullptr, tbuf,  NTF, 256, 128, 1);
  aconv(tbuf,  bx_scr, (size_t)NTF * 256);
  gemm(stream, bx_scr, headWT[2], headB[2], nullptr, hs,    NTF, 384, 256, 1);
  aconv(hs,    bx_scr, (size_t)NTF * 384);
  gemm(stream, bx_scr, headWT[3], headB[3], nullptr, h_tf,  NTF, 256, 384, 0);
}
